// TemporalAttention_43722767073503
// MI455X (gfx1250) — matrix operand order repaired and checked by reading, not compiled, not run on hardware
//
#include <hip/hip_runtime.h>

// CDNA5 / gfx1250, wave32. WMMA vector operand types (probe-verified).
typedef __attribute__((ext_vector_type(16))) _Float16 v16h;
typedef __attribute__((ext_vector_type(8)))  float    v8f;

#define B_ 256
#define T_ 2048
#define H_ 128

// ---------------------------------------------------------------------------
// Prep 1: pack w1 (H x H, stored (k_in, n_out) row-major, fp32) into fp16
// WMMA B-fragments, laid out exactly per-lane so the main kernel loads each
// fragment with one 32-byte vector load.
// Fragment id: tile = nt*4 + ks   (nt = n-tile 0..7, ks = k-step 0..3)
// B layout (16x16x32 f16): n = lane%16 ; k = ks*32 + (j<8 ? 0 : 16) + (lane/16)*8 + (j%8)
// ---------------------------------------------------------------------------
__global__ void pack_w1_kernel(const float* __restrict__ w1,
                               _Float16* __restrict__ w1h) {
  const int tile = blockIdx.x;    // 0..31
  const int lane = threadIdx.x;   // 0..31
  const int nt = tile >> 2, ks = tile & 3;
  const int n = nt * 16 + (lane & 15);
  const int kbase = ks * 32 + (lane >> 4) * 8;
  _Float16* dst = w1h + ((size_t)tile * 32 + (size_t)lane) * 16;
#pragma unroll
  for (int j = 0; j < 16; ++j)
    dst[j] = (_Float16)w1[(size_t)(kbase + (j < 8 ? j : j + 8)) * H_ + n];
}

// ---------------------------------------------------------------------------
// Prep 2: qbias[b][o] = b1[o] + b2[o] + concat(h[b],c[b]) . w2[:,o]
// query is broadcast over T in the reference, so this hoists the whole
// second GEMM out of the T loop (O(B*H*2H) once instead of per-timestep).
// ---------------------------------------------------------------------------
__global__ void qbias_kernel(const float* __restrict__ h,
                             const float* __restrict__ c,
                             const float* __restrict__ w2,
                             const float* __restrict__ b1,
                             const float* __restrict__ b2,
                             float* __restrict__ qbias) {
  const int b = blockIdx.x;    // 0..255
  const int o = threadIdx.x;   // 0..127
  const float* hb = h + (size_t)b * H_;
  const float* cb = c + (size_t)b * H_;
  float acc = b1[o] + b2[o];
  for (int g = 0; g < H_; ++g) acc = fmaf(hb[g], w2[(size_t)g * H_ + o], acc);
  for (int g = 0; g < H_; ++g) acc = fmaf(cb[g], w2[(size_t)(H_ + g) * H_ + o], acc);
  qbias[(size_t)b * H_ + o] = acc;
}

// gfx1250 has a hardware V_TANH_F32 (single TRANS op, co-executes with the
// WMMA pipe). Fall back to a 3-op exp2/rcp form that preserves NaN/Inf
// semantics (NaN->NaN, +Inf->1, -Inf->-1) if the builtin is unavailable.
__device__ __forceinline__ float fast_tanh(float x) {
#if __has_builtin(__builtin_amdgcn_tanhf)
  return __builtin_amdgcn_tanhf(x);
#else
  float e = __builtin_amdgcn_exp2f(x * 2.88539008177792681f);
  return 1.0f - 2.0f * __builtin_amdgcn_rcpf(e + 1.0f);
#endif
}

// Butterfly sum across each 16-lane half-wave (the N-group of the WMMA C
// layout) using ds_swizzle group-of-32 XOR mode: offset = xor<<10 | 0x1f.
__device__ __forceinline__ float lane16_sum(float s) {
  s += __int_as_float(__builtin_amdgcn_ds_swizzle(__float_as_int(s), 0x041F)); // xor 1
  s += __int_as_float(__builtin_amdgcn_ds_swizzle(__float_as_int(s), 0x081F)); // xor 2
  s += __int_as_float(__builtin_amdgcn_ds_swizzle(__float_as_int(s), 0x101F)); // xor 4
  s += __int_as_float(__builtin_amdgcn_ds_swizzle(__float_as_int(s), 0x201F)); // xor 8
  return s;
}

// ---------------------------------------------------------------------------
// Main: 256 threads = 8 waves per block; w1h (32 KiB of fp16 B-fragments) is
// staged once per workgroup into LDS via the gfx1250 async global->LDS path,
// then all 8 waves consume it with ds_load_b128 (global B traffic: 128 MB
// total instead of 1 GB; B latency off the L2 round-trip path). Each wave
// computes a 16-row x 128-col tile of score = enc_h @ w1 (+qbias), tanh,
// dot with v, then out = s - s (log_softmax over a singleton axis).
// ---------------------------------------------------------------------------
__global__ __launch_bounds__(256) void attn_score_kernel(
    const float* __restrict__ enc_h, const _Float16* __restrict__ w1h,
    const float* __restrict__ qbias, const float* __restrict__ vvec,
    const float* __restrict__ bv, float* __restrict__ out) {
  __shared__ _Float16 sw1[32 * 32 * 16];   // 32 KiB: all 32 B-fragment tiles

  const int tid  = threadIdx.x;
  const int lane = tid & 31;
  const int wv   = tid >> 5;           // wave 0..7 -> 16-row sub-tile
  const int g    = lane >> 4;          // lane group (selects K-halves / M+8)
  const int l16  = lane & 15;
  const long rowBase = (long)blockIdx.x * 128 + (long)wv * 16;
  const int  b = (int)(blockIdx.x >> 4);   // 16 blocks per batch (2048/128)

  // ---- stage w1h -> LDS (32768 B, 128 B per thread) ----
#if __has_builtin(__builtin_amdgcn_global_load_async_to_lds_b128)
  {
    typedef int v4i128 __attribute__((vector_size(16)));
    typedef __attribute__((address_space(1))) v4i128 as1_v4i;
    typedef __attribute__((address_space(3))) v4i128 as3_v4i;
    as1_v4i* gsrc = (as1_v4i*)w1h;   // global src, 16B elements
    as3_v4i* ldst = (as3_v4i*)sw1;   // LDS dst, 16B elements
#pragma unroll
    for (int i = 0; i < 8; ++i) {
      const int idx = i * 256 + tid;
      __builtin_amdgcn_global_load_async_to_lds_b128(gsrc + idx, ldst + idx, 0, 0);
    }
#if __has_builtin(__builtin_amdgcn_s_wait_asynccnt)
    __builtin_amdgcn_s_wait_asynccnt(0);
#else
    asm volatile("s_wait_asynccnt 0x0" ::: "memory");
#endif
  }
#else
  {
    const float4* gsrc = (const float4*)w1h;
    float4* ldst = (float4*)sw1;
#pragma unroll
    for (int i = 0; i < 8; ++i) ldst[i * 256 + tid] = gsrc[i * 256 + tid];
  }
#endif
  __syncthreads();

  // A-fragment source row: lanes 0-15 hold M=lane, lanes 16-31 hold M=lane-16.
  const float* arow = enc_h + (rowBase + l16) * (long)H_ + g * 8;

  v8f acc[8];
#pragma unroll
  for (int nt = 0; nt < 8; ++nt) { v8f z = {}; acc[nt] = z; }

#pragma unroll
  for (int ks = 0; ks < 4; ++ks) {
    // A 16x32 f16 layout: half j -> k = ks*32 + (j<8 ? 0 : 16) + g*8 + (j%8)
    const float* ap = arow + ks * 32;
    float4 a0 = *(const float4*)(ap + 0);
    float4 a1 = *(const float4*)(ap + 4);
    float4 a2 = *(const float4*)(ap + 16);
    float4 a3 = *(const float4*)(ap + 20);
    v16h af;
    af[0]  = (_Float16)a0.x; af[1]  = (_Float16)a0.y;
    af[2]  = (_Float16)a0.z; af[3]  = (_Float16)a0.w;
    af[4]  = (_Float16)a1.x; af[5]  = (_Float16)a1.y;
    af[6]  = (_Float16)a1.z; af[7]  = (_Float16)a1.w;
    af[8]  = (_Float16)a2.x; af[9]  = (_Float16)a2.y;
    af[10] = (_Float16)a2.z; af[11] = (_Float16)a2.w;
    af[12] = (_Float16)a3.x; af[13] = (_Float16)a3.y;
    af[14] = (_Float16)a3.z; af[15] = (_Float16)a3.w;

#pragma unroll
    for (int nt = 0; nt < 8; ++nt) {
      v16h bf = *(const v16h*)(sw1 + ((nt * 4 + ks) * 32 + lane) * 16);
      acc[nt] = __builtin_amdgcn_wmma_f32_16x16x32_f16(
          /*neg_a=*/false, af, /*neg_b=*/false, bf,
          /*c_mod=*/(short)0, acc[nt], /*reuse_a=*/false, /*reuse_b=*/false);
    }
  }

  // Epilogue: C layout — VGPR r holds row m = r + g*8, column n = l16 + nt*16.
  float rowacc[8];
#pragma unroll
  for (int r = 0; r < 8; ++r) rowacc[r] = 0.f;
#pragma unroll
  for (int nt = 0; nt < 8; ++nt) {
    const int n = nt * 16 + l16;
    const float qb = qbias[(size_t)b * H_ + n];
    const float vn = vvec[n];
#pragma unroll
    for (int r = 0; r < 8; ++r)
      rowacc[r] = fmaf(fast_tanh(acc[nt][r] + qb), vn, rowacc[r]);
  }
  const float bvv = bv[0];
#pragma unroll
  for (int r = 0; r < 8; ++r) {
    float s = lane16_sum(rowacc[r]) + bvv;
    float res = s - s;   // log_softmax over axis of size 1: 0, NaN-propagating
    if (l16 == 0) out[rowBase + g * 8 + r] = res;
  }
}

extern "C" void kernel_launch(void* const* d_in, const int* in_sizes, int n_in,
                              void* d_out, int out_size, void* d_ws, size_t ws_size,
                              hipStream_t stream) {
  (void)in_sizes; (void)n_in; (void)out_size; (void)ws_size;
  const float* enc_h = (const float*)d_in[0];
  const float* h     = (const float*)d_in[1];
  const float* c     = (const float*)d_in[2];
  const float* w1    = (const float*)d_in[3];
  const float* b1    = (const float*)d_in[4];
  const float* w2    = (const float*)d_in[5];
  const float* b2    = (const float*)d_in[6];
  const float* v     = (const float*)d_in[7];
  const float* bv    = (const float*)d_in[8];
  float* out = (float*)d_out;

  _Float16* w1h   = (_Float16*)d_ws;                    // 32 KiB fp16 B-fragments
  float*    qbias = (float*)((char*)d_ws + 32 * 1024);  // 128 KiB per-batch bias

  hipLaunchKernelGGL(pack_w1_kernel, dim3(32), dim3(32), 0, stream, w1, w1h);
  hipLaunchKernelGGL(qbias_kernel, dim3(B_), dim3(H_), 0, stream,
                     h, c, w2, b1, b2, qbias);
  hipLaunchKernelGGL(attn_score_kernel, dim3((B_ * T_) / 128), dim3(256), 0, stream,
                     enc_h, w1h, qbias, v, bv, out);
}
